// MultiHeadAttention_83468394430720
// MI455X (gfx1250) — compile-verified, matrix operand order checked by reading, not run on hardware
//
#include <hip/hip_runtime.h>

#define EMB   512
#define HEADS 8
#define HD    64
#define BATCH 32
#define SEQ   4096
#define TOK   (BATCH * SEQ)   // 131072
#define QK_CK 128             // token-chunk staged in LDS by k_qk

typedef _Float16 h16;
typedef __attribute__((ext_vector_type(16))) _Float16 v16h;
typedef __attribute__((ext_vector_type(8)))  _Float16 v8h;
typedef __attribute__((ext_vector_type(8)))  float    v8f;
typedef __attribute__((ext_vector_type(4)))  float    v4f;
typedef __attribute__((ext_vector_type(4)))  int      v4i;

union V16H { v16h v; v8h h2[2]; h16 e[16]; };
union V8F  { v8f  v; float e[8]; };
union V8H  { v8h  v; h16 e[8]; };

#define AS1 __attribute__((address_space(1)))
#define AS3 __attribute__((address_space(3)))

#if defined(__has_builtin)
#if __has_builtin(__builtin_amdgcn_global_load_async_to_lds_b128) && \
    __has_builtin(__builtin_amdgcn_s_wait_asynccnt)
#define ASYNC_LDS 1
#endif
#endif
#ifndef ASYNC_LDS
#define ASYNC_LDS 0
#endif

// ---------------------------------------------------------------------------
// WMMA helpers (v_wmma_f32_16x16x32_f16, wave32)
// A-frag (16x32, row-major source): lane = {m = lane&15, khalf = lane>>4}
//   elems [0..7]  = M[m][k0 + 8*khalf + i]
//   elems [8..15] = M[m][k0 + 8*khalf + 16 + i]
// B-frag (32x16) from a row-major N x K matrix uses the same loader
// (lane&15 indexes the output column = matrix row).
// ---------------------------------------------------------------------------
__device__ inline v16h frag_ld_f16(const h16* __restrict__ p, int ld) {
  const int lane = threadIdx.x & 31;
  const h16* q = p + (size_t)(lane & 15) * ld + ((lane >> 4) << 3);
  V16H r;
  r.h2[0] = *(const v8h*)(q);
  r.h2[1] = *(const v8h*)(q + 16);
  return r.v;
}

__device__ inline v8f wmma16(v16h a, v16h b, v8f c) {
  return __builtin_amdgcn_wmma_f32_16x16x32_f16(false, a, false, b,
                                                (short)0, c, false, false);
}

// ---------------------------------------------------------------------------
// Streaming f32 -> f16 convert for x (keeps the GEMM inner loop pure WMMA).
// ---------------------------------------------------------------------------
__global__ __launch_bounds__(256)
void k_convert(const float* __restrict__ x, h16* __restrict__ xh)
{
  const size_t i = ((size_t)blockIdx.x * 256 + threadIdx.x) * 8;
  v4f a = *(const v4f*)(x + i);
  v4f b = *(const v4f*)(x + i + 4);
  V8H o;
#pragma unroll
  for (int j = 0; j < 4; ++j) {
    o.e[j]     = (h16)a[j];
    o.e[4 + j] = (h16)b[j];
  }
  *(v8h*)(xh + i) = o.v;
}

// ---------------------------------------------------------------------------
// Pack weights to f16: Wqkv (1536 x 512, rows = output channel), Wo (512x512),
// concatenated bias (1536) in f32.
// ---------------------------------------------------------------------------
__global__ __launch_bounds__(256)
void k_pack(const float* __restrict__ wq, const float* __restrict__ wk,
            const float* __restrict__ wv, const float* __restrict__ wo,
            const float* __restrict__ bq, const float* __restrict__ bk,
            const float* __restrict__ bv,
            h16* __restrict__ wqkv, h16* __restrict__ woh,
            float* __restrict__ bqkv)
{
  const int i = blockIdx.x * 256 + threadIdx.x;
  const int WQKV = 1536 * EMB;
  const int WO   = EMB * EMB;
  if (i < WQKV) {
    int j = i / EMB, e = i % EMB;
    float v = (j < 512)  ? wq[j * EMB + e]
            : (j < 1024) ? wk[(j - 512) * EMB + e]
                         : wv[(j - 1024) * EMB + e];
    wqkv[i] = (h16)v;
  } else if (i < WQKV + WO) {
    int t = i - WQKV;
    woh[t] = (h16)wo[t];
  } else if (i < WQKV + WO + 1536) {
    int j = i - WQKV - WO;
    bqkv[j] = (j < 512) ? bq[j] : (j < 1024) ? bk[j - 512] : bv[j - 1024];
  }
}

// ---------------------------------------------------------------------------
// Fused QKV projection: C(t, j) = sum_e xh(t,e) * Wqkv(j,e) + b(j), j in [0,1536)
// Epilogue scatters: Q,K -> transposed (bh*64+d, n) f16 buffers (contiguous
// v8h stores), V -> token-major. Block 256 thr = 8 waves; wave tile 32x32.
// ---------------------------------------------------------------------------
__global__ __launch_bounds__(256)
void k_qkv_gemm(const h16* __restrict__ xh,      // (TOK, EMB) f16
                const h16* __restrict__ wqkv,    // (1536, EMB) f16
                const float* __restrict__ bqkv,  // (1536)
                h16* __restrict__ qt,            // (BATCH*HEADS*HD, SEQ)
                h16* __restrict__ kt,            // (BATCH*HEADS*HD, SEQ)
                h16* __restrict__ vh)            // (TOK, EMB)
{
  const int t0   = blockIdx.x * 128;
  const int n0   = blockIdx.y * 64;
  const int wave = threadIdx.x >> 5;
  const int lane = threadIdx.x & 31;
  const int wm = (wave & 3) * 32;
  const int wn = (wave >> 2) * 32;

  const h16* xp0 = xh + (size_t)(t0 + wm) * EMB;
  const h16* xp1 = xh + (size_t)(t0 + wm + 16) * EMB;
  const h16* wp0 = wqkv + (size_t)(n0 + wn) * EMB;
  const h16* wp1 = wqkv + (size_t)(n0 + wn + 16) * EMB;

  v8f acc[2][2] = {};
  for (int k0 = 0; k0 < EMB; k0 += 32) {
    __builtin_prefetch(xp0 + k0 + 128, 0, 1);   // next A tiles -> L2/L0
    __builtin_prefetch(xp1 + k0 + 128, 0, 1);
    v16h a0 = frag_ld_f16(xp0 + k0, EMB);
    v16h a1 = frag_ld_f16(xp1 + k0, EMB);
    v16h b0 = frag_ld_f16(wp0 + k0, EMB);
    v16h b1 = frag_ld_f16(wp1 + k0, EMB);
    acc[0][0] = wmma16(a0, b0, acc[0][0]);
    acc[0][1] = wmma16(a0, b1, acc[0][1]);
    acc[1][0] = wmma16(a1, b0, acc[1][0]);
    acc[1][1] = wmma16(a1, b1, acc[1][1]);
  }

  const int b = t0 / SEQ;   // 128 | 4096 -> block lies in one batch
#pragma unroll
  for (int mi = 0; mi < 2; ++mi)
#pragma unroll
    for (int ni = 0; ni < 2; ++ni) {
      const int j    = n0 + wn + ni * 16 + (lane & 15);          // 0..1535
      const float bias = bqkv[j];
      const int mrow = t0 + wm + mi * 16 + ((lane >> 4) << 3);   // token of e[0]
      V8F a; a.v = acc[mi][ni];
      if (j < 1024) {                       // Q or K -> transposed layout
        h16* dst = (j < 512) ? qt : kt;
        const int jj = j & 511;
        const size_t row = (size_t)(b * HEADS + (jj >> 6)) * HD + (jj & 63);
        const int col = mrow - b * SEQ;
        V8H o;
#pragma unroll
        for (int v = 0; v < 8; ++v) o.e[v] = (h16)(a.e[v] + bias);
        *(v8h*)(dst + row * SEQ + col) = o.v;
      } else {                              // V -> token-major
        const int d = j - 1024;             // 0..511 (= h*64 + dv)
#pragma unroll
        for (int v = 0; v < 8; ++v)
          vh[(size_t)(mrow + v) * EMB + d] = (h16)(a.e[v] + bias);
      }
    }
}

// ---------------------------------------------------------------------------
// Scores: S(bh, dq, dk) = (1/8) * sum_n Q^T(dq,n) * K(n,dk).
// One block per (b,h): 512 thr = 16 waves, one 16x16 tile each; K = 4096.
// Q/K chunks are staged in LDS (async-to-LDS double buffer when available);
// each LDS row is then consumed by 4 waves -> 4x global-traffic reduction.
// ---------------------------------------------------------------------------
__global__ __launch_bounds__(512)
void k_qk(const h16* __restrict__ qt, const h16* __restrict__ kt,
          float* __restrict__ s)
{
  __shared__ h16 qs[2][HD][QK_CK];
  __shared__ h16 ks[2][HD][QK_CK];

  const int bh   = blockIdx.x;          // 0..255
  const int tid  = threadIdx.x;
  const int wave = tid >> 5;            // 0..15
  const int lane = tid & 31;
  const int mt = (wave & 3) * 16;
  const int nt = (wave >> 2) * 16;

  // Staging map: thread -> (row = tid>>3, 16 halves at col (tid&7)*16).
  const int srow = tid >> 3;            // 0..63
  const int scol = (tid & 7) * 16;      // halves
  const h16* qg = qt + (size_t)(bh * HD + srow) * SEQ + scol;
  const h16* kg = kt + (size_t)(bh * HD + srow) * SEQ + scol;

  auto stage = [&](int bufi, int c0) {
#if ASYNC_LDS
    __builtin_amdgcn_global_load_async_to_lds_b128(
        (AS1 v4i*)(qg + c0),     (AS3 v4i*)&qs[bufi][srow][scol], 0, 0);
    __builtin_amdgcn_global_load_async_to_lds_b128(
        (AS1 v4i*)(qg + c0 + 8), (AS3 v4i*)&qs[bufi][srow][scol + 8], 0, 0);
    __builtin_amdgcn_global_load_async_to_lds_b128(
        (AS1 v4i*)(kg + c0),     (AS3 v4i*)&ks[bufi][srow][scol], 0, 0);
    __builtin_amdgcn_global_load_async_to_lds_b128(
        (AS1 v4i*)(kg + c0 + 8), (AS3 v4i*)&ks[bufi][srow][scol + 8], 0, 0);
#else
    *(v8h*)&qs[bufi][srow][scol]     = *(const v8h*)(qg + c0);
    *(v8h*)&qs[bufi][srow][scol + 8] = *(const v8h*)(qg + c0 + 8);
    *(v8h*)&ks[bufi][srow][scol]     = *(const v8h*)(kg + c0);
    *(v8h*)&ks[bufi][srow][scol + 8] = *(const v8h*)(kg + c0 + 8);
#endif
  };

  stage(0, 0);

  v8f acc = {};
  const int NCHUNK = SEQ / QK_CK;       // 32
  for (int c = 0; c < NCHUNK; ++c) {
    const int buf = c & 1;
    // Issue next chunk (wraps harmlessly on the last iteration), then wait
    // for the current chunk's 4 per-thread async loads (in-order completion).
    stage(buf ^ 1, ((c + 1) & (NCHUNK - 1)) * QK_CK);
#if ASYNC_LDS
    __builtin_amdgcn_s_wait_asynccnt(4);
#endif
    __syncthreads();

    const h16* ap = &qs[buf][mt][0];
    const h16* bp = &ks[buf][nt][0];
#pragma unroll
    for (int k0 = 0; k0 < QK_CK; k0 += 32) {
      v16h a = frag_ld_f16(ap + k0, QK_CK);
      v16h b = frag_ld_f16(bp + k0, QK_CK);
      acc = wmma16(a, b, acc);
    }
    __syncthreads();
  }

  const float scale = 0.125f;           // 1/sqrt(HD)
  V8F o; o.v = acc;
  const int n  = nt + (lane & 15);
  const int mb = mt + ((lane >> 4) << 3);
  float* srow_out = s + (size_t)bh * HD * HD;
#pragma unroll
  for (int v = 0; v < 8; ++v)
    srow_out[(size_t)(mb + v) * HD + n] = o.e[v] * scale;
}

// ---------------------------------------------------------------------------
// Softmax over dk (64-wide) per (bh, dq) row; output f16 row-major attn.
// ---------------------------------------------------------------------------
__global__ __launch_bounds__(64)
void k_softmax(const float* __restrict__ s, h16* __restrict__ ah)
{
  const int bh = blockIdx.x;
  const int q  = threadIdx.x;
  const float* row = s + ((size_t)bh * HD + q) * HD;
  float m = -3.4e38f;
  for (int i = 0; i < HD; ++i) m = fmaxf(m, row[i]);
  float sum = 0.f;
  for (int i = 0; i < HD; ++i) sum += __expf(row[i] - m);
  const float inv = 1.f / sum;
  h16* orow = ah + ((size_t)bh * HD + q) * HD;
  for (int i = 0; i < HD; ++i) orow[i] = (h16)(__expf(row[i] - m) * inv);
}

// ---------------------------------------------------------------------------
// weighted(t, h*64+q) = sum_k V(t, h*64+k) * attn(bh, q, k)   [= V x attn^T]
// Grid: (SEQ/128, bh). Block 256 = 8 waves; wave: 16 tokens x 64 cols, K=64.
// ---------------------------------------------------------------------------
__global__ __launch_bounds__(256)
void k_weighted(const h16* __restrict__ vh, const h16* __restrict__ ah,
                h16* __restrict__ wout)
{
  const int bh = blockIdx.y;
  const int b = bh >> 3, h = bh & 7;
  const int wave = threadIdx.x >> 5;
  const int lane = threadIdx.x & 31;
  const int t0 = b * SEQ + blockIdx.x * 128 + wave * 16;  // global token base
  const h16* vp = vh + (size_t)t0 * EMB + h * HD;
  const h16* ap = ah + (size_t)bh * HD * HD;

  v8f acc[4] = {};
#pragma unroll
  for (int k0 = 0; k0 < HD; k0 += 32) {
    v16h a = frag_ld_f16(vp + k0, EMB);
#pragma unroll
    for (int nt = 0; nt < 4; ++nt) {
      v16h bf = frag_ld_f16(ap + (size_t)(nt * 16) * HD + k0, HD);
      acc[nt] = wmma16(a, bf, acc[nt]);
    }
  }

  const int mb = (lane >> 4) << 3;
#pragma unroll
  for (int nt = 0; nt < 4; ++nt) {
    const int qcol = nt * 16 + (lane & 15);
    V8F o; o.v = acc[nt];
#pragma unroll
    for (int v = 0; v < 8; ++v)
      wout[(size_t)(t0 + mb + v) * EMB + h * HD + qcol] = (h16)o.e[v];
  }
}

// ---------------------------------------------------------------------------
// Output projection: out(t, j) = sum_e weighted(t,e) * Wo(j,e) + bo(j), f32 out.
// ---------------------------------------------------------------------------
__global__ __launch_bounds__(256)
void k_outproj(const h16* __restrict__ wout, const h16* __restrict__ woh,
               const float* __restrict__ bo, float* __restrict__ out)
{
  const int t0   = blockIdx.x * 128;
  const int n0   = blockIdx.y * 64;
  const int wave = threadIdx.x >> 5;
  const int lane = threadIdx.x & 31;
  const int wm = (wave & 3) * 32;
  const int wn = (wave >> 2) * 32;

  const h16* ap0 = wout + (size_t)(t0 + wm) * EMB;
  const h16* ap1 = wout + (size_t)(t0 + wm + 16) * EMB;
  const h16* bp0 = woh + (size_t)(n0 + wn) * EMB;
  const h16* bp1 = woh + (size_t)(n0 + wn + 16) * EMB;

  v8f acc[2][2] = {};
  for (int k0 = 0; k0 < EMB; k0 += 32) {
    __builtin_prefetch(ap0 + k0 + 128, 0, 1);
    __builtin_prefetch(ap1 + k0 + 128, 0, 1);
    v16h a0 = frag_ld_f16(ap0 + k0, EMB);
    v16h a1 = frag_ld_f16(ap1 + k0, EMB);
    v16h b0 = frag_ld_f16(bp0 + k0, EMB);
    v16h b1 = frag_ld_f16(bp1 + k0, EMB);
    acc[0][0] = wmma16(a0, b0, acc[0][0]);
    acc[0][1] = wmma16(a0, b1, acc[0][1]);
    acc[1][0] = wmma16(a1, b0, acc[1][0]);
    acc[1][1] = wmma16(a1, b1, acc[1][1]);
  }

#pragma unroll
  for (int mi = 0; mi < 2; ++mi)
#pragma unroll
    for (int ni = 0; ni < 2; ++ni) {
      const int j  = n0 + wn + ni * 16 + (lane & 15);
      const float bias = bo[j];
      const int m0 = t0 + wm + mi * 16 + ((lane >> 4) << 3);
      V8F o; o.v = acc[mi][ni];
#pragma unroll
      for (int v = 0; v < 8; ++v)
        out[(size_t)(m0 + v) * EMB + j] = o.e[v] + bias;
    }
}

// ---------------------------------------------------------------------------
extern "C" void kernel_launch(void* const* d_in, const int* in_sizes, int n_in,
                              void* d_out, int out_size, void* d_ws, size_t ws_size,
                              hipStream_t stream)
{
  const float* x  = (const float*)d_in[0];
  const float* wq = (const float*)d_in[1];
  const float* bq = (const float*)d_in[2];
  const float* wk = (const float*)d_in[3];
  const float* bk = (const float*)d_in[4];
  const float* wv = (const float*)d_in[5];
  const float* bv = (const float*)d_in[6];
  const float* wo = (const float*)d_in[7];
  const float* bo = (const float*)d_in[8];
  (void)in_sizes; (void)n_in; (void)out_size; (void)ws_size;

  char* ws = (char*)d_ws;
  size_t off = 0;
  auto alloc = [&](size_t bytes) {
    char* p = ws + off;
    off = (off + bytes + 255) & ~(size_t)255;
    return p;
  };
  h16*   xh   = (h16*)  alloc((size_t)TOK * EMB * 2);
  h16*   wqkv = (h16*)  alloc((size_t)1536 * EMB * 2);
  h16*   woh  = (h16*)  alloc((size_t)EMB * EMB * 2);
  float* bqkv = (float*)alloc(1536 * 4);
  h16*   qt   = (h16*)  alloc((size_t)BATCH * HEADS * HD * SEQ * 2);
  h16*   kt   = (h16*)  alloc((size_t)BATCH * HEADS * HD * SEQ * 2);
  h16*   vh   = (h16*)  alloc((size_t)TOK * EMB * 2);
  float* s    = (float*)alloc((size_t)BATCH * HEADS * HD * HD * 4);
  h16*   ah   = (h16*)  alloc((size_t)BATCH * HEADS * HD * HD * 2);
  h16*   wout = (h16*)  alloc((size_t)TOK * EMB * 2);

  k_convert<<<dim3((TOK * EMB) / (8 * 256)), 256, 0, stream>>>(x, xh);

  const int packN = 1536 * EMB + EMB * EMB + 1536;
  k_pack<<<dim3((packN + 255) / 256), 256, 0, stream>>>(
      wq, wk, wv, wo, bq, bk, bv, wqkv, woh, bqkv);

  k_qkv_gemm<<<dim3(TOK / 128, 1536 / 64), 256, 0, stream>>>(
      xh, wqkv, bqkv, qt, kt, vh);

  k_qk<<<dim3(BATCH * HEADS), 512, 0, stream>>>(qt, kt, s);

  k_softmax<<<dim3(BATCH * HEADS), 64, 0, stream>>>(s, ah);

  k_weighted<<<dim3(SEQ / 128, BATCH * HEADS), 256, 0, stream>>>(vh, ah, wout);

  k_outproj<<<dim3(TOK / 128, EMB / 64), 256, 0, stream>>>(
      wout, woh, bo, (float*)d_out);
}
